// _GeodesicSelfAttention_644245095062
// MI455X (gfx1250) — matrix operand order repaired and checked by reading, not compiled, not run on hardware
//
#include <hip/hip_runtime.h>
#include <hip/hip_bf16.h>

typedef __attribute__((ext_vector_type(16))) _Float16 v16h;
typedef __attribute__((ext_vector_type(8)))  _Float16 v8h;
typedef __attribute__((ext_vector_type(8)))  float    v8f;

static constexpr int Bq = 2;
static constexpr int T  = 2048;
static constexpr int D  = 1024;
static constexpr int H  = 16;
static constexpr int DH = 64;

// ---------------- WMMA helpers (layouts per CDNA5 ISA 7.12.2) ---

__device__ __forceinline__ v8f wmma_f16(v16h a, v16h b, v8f c) {
  // v_wmma_f32_16x16x32_f16 : D = A(16x32 f16) * B(32x16 f16) + C(16x16 f32)
  return __builtin_amdgcn_wmma_f32_16x16x32_f16(
      /*neg_a=*/false, a, /*neg_b=*/false, b,
      /*c_mod=*/(short)0, c, /*reuse_a=*/false, /*reuse_b=*/false);
}

// A-fragment: lane l<16 holds row l, K = {0..7, 16..23}; lane>=16 row l-16,
// K = {8..15, 24..31}.  'p' must already point at  [row0][k0 + 8*(lane>>4)].
__device__ __forceinline__ v16h load_a16(const _Float16* p) {
  v8h lo = *(const v8h*)p;         // K base..base+7
  v8h hi = *(const v8h*)(p + 16);  // K base+16..base+23
  return __builtin_shufflevector(lo, hi, 0,1,2,3,4,5,6,7,8,9,10,11,12,13,14,15);
}

// CDNA5 async copy: global -> LDS, 16B per lane, tracked by ASYNCcnt.
__device__ __forceinline__ void async_b128_to_lds(const void* gptr, void* lptr) {
  unsigned lds          = (unsigned)(uintptr_t)lptr;  // low 32b of flat LDS ptr
  unsigned long long ga = (unsigned long long)(uintptr_t)gptr;
  asm volatile("global_load_async_to_lds_b128 %0, %1, off"
               :: "v"(lds), "v"(ga) : "memory");
}
__device__ __forceinline__ void wait_async0() {
  asm volatile("s_wait_asynccnt 0x0" ::: "memory");
}

// 16x16 GEMM tile: C = A[16xK] * Bt[16xK]^T  (Bt stores B transposed: row n = col n)
__device__ __forceinline__ v8f mm_tile(const _Float16* A, int lda,
                                       const _Float16* Bt, int ldb,
                                       int K, int lane) {
  const int l = lane & 15, half = lane >> 4;
  const _Float16* ap = A  + (size_t)l * lda + 8 * half;
  const _Float16* bp = Bt + (size_t)l * ldb + 8 * half;
  v8f c = {};
  for (int k = 0; k < K; k += 32) {
    v16h a = load_a16(ap + k);
    v16h b = load_a16(bp + k);   // B: lanes 0-15 K={0..7,16..23}, lanes 16-31 K={8..15,24..31}
    c = wmma_f16(a, b, c);
  }
  return c;
}

// ---------------- kernel 1: cast / transpose / diag prep ------------------

__global__ void __launch_bounds__(256)
prep_kernel(const float* __restrict__ x,
            const float* __restrict__ Wq, const float* __restrict__ Wk,
            const float* __restrict__ Wv, const float* __restrict__ Wo,
            const float* __restrict__ log_diag,
            _Float16* __restrict__ xh,
            _Float16* __restrict__ Wqt, _Float16* __restrict__ Wkt,
            _Float16* __restrict__ Wvt, _Float16* __restrict__ Wot,
            float* __restrict__ sd) {
  int i = blockIdx.x * blockDim.x + threadIdx.x;     // 0 .. B*T*D-1 (4M)
  xh[i] = (_Float16)x[i];
  if (i < D * D) {
    int n = i / D, k = i % D;                        // Wt[n][k] = W[k][n]
    Wqt[(size_t)n * D + k] = (_Float16)Wq[(size_t)k * D + n];
    Wkt[(size_t)n * D + k] = (_Float16)Wk[(size_t)k * D + n];
    Wvt[(size_t)n * D + k] = (_Float16)Wv[(size_t)k * D + n];
    Wot[(size_t)n * D + k] = (_Float16)Wo[(size_t)k * D + n];
  }
  if (i < D) {
    float ld = log_diag[i];
    float sp = (ld > 20.f) ? ld : log1pf(__expf(ld));
    sd[i] = sqrtf(sp + 1e-6f);                       // fold sqrt(diag) into q,k
  }
}

// ---------------- kernel 2: QKV projection (WMMA) -------------------------
// out = (x @ W + b) * (q,k: sqrt(diag) ; v: 1)  stored f16 in [B*T, D] layout

__global__ void __launch_bounds__(256)
qkv_kernel(const _Float16* __restrict__ xh,
           const _Float16* __restrict__ Wqt, const _Float16* __restrict__ Wkt,
           const _Float16* __restrict__ Wvt,
           const float* __restrict__ bqv, const float* __restrict__ bkv,
           const float* __restrict__ bvv, const float* __restrict__ sd,
           _Float16* __restrict__ qs, _Float16* __restrict__ ks,
           _Float16* __restrict__ vv) {
  const int wave = (blockIdx.x * blockDim.x + threadIdx.x) >> 5;
  const int lane = threadIdx.x & 31;
  const int tilesPerMat = (Bq * T / 16) * (D / 16);  // 256*64
  const int sel = wave / tilesPerMat;
  const int tt  = wave % tilesPerMat;
  const int mt = tt / (D / 16), nt = tt % (D / 16);

  const _Float16* Wt = (sel == 0) ? Wqt : (sel == 1) ? Wkt : Wvt;
  const float* bias  = (sel == 0) ? bqv : (sel == 1) ? bkv : bvv;
  _Float16* out      = (sel == 0) ? qs  : (sel == 1) ? ks  : vv;

  v8f c = mm_tile(xh + (size_t)mt * 16 * D, D, Wt + (size_t)nt * 16 * D, D, D, lane);

  const int l = lane & 15, half = lane >> 4;
  const int n = nt * 16 + l;
  const float bb = bias[n];
  const float ss = (sel < 2) ? sd[n] : 1.0f;
#pragma unroll
  for (int r = 0; r < 8; ++r) {
    int m = mt * 16 + r + 8 * half;                  // C layout: VGPR r -> row r / r+8
    out[(size_t)m * D + n] = (_Float16)((c[r] + bb) * ss);
  }
}

// ---------------- kernel 3: per-key |ks|^2 --------------------------------

__global__ void __launch_bounds__(256)
kss_kernel(const _Float16* __restrict__ ks, float* __restrict__ kss) {
  int i = blockIdx.x * blockDim.x + threadIdx.x;     // 0 .. B*H*T-1
  if (i >= Bq * H * T) return;
  int t = i % T, bh = i / T, b = bh / H, h = bh % H;
  const _Float16* p = ks + ((size_t)(b * T + t)) * D + h * DH;
  float s = 0.f;
#pragma unroll
  for (int j8 = 0; j8 < DH / 8; ++j8) {
    v8h v = *(const v8h*)(p + j8 * 8);
#pragma unroll
    for (int e = 0; e < 8; ++e) { float f = (float)v[e]; s += f * f; }
  }
  kss[i] = s;
}

// ---------------- kernel 4: flash attention (WMMA + async LDS copy) -------
// scores = 0.0625*(qs.ks) - 0.03125*|ks|^2 + (mask?0:-1e30) ; online softmax

__global__ void __launch_bounds__(128)
attn_kernel(const _Float16* __restrict__ qs, const _Float16* __restrict__ ks,
            const _Float16* __restrict__ vv, const float* __restrict__ kssp,
            const unsigned char* __restrict__ mask, _Float16* __restrict__ ah) {
  __shared__ _Float16 Ks[2][32 * 64];   // double-buffered key block [s][dh]
  __shared__ _Float16 Vt[64 * 32];      // value block transposed [dh][s]
  __shared__ _Float16 Pl[4][16 * 32];   // per-wave P transpose buffer

  const int tid  = threadIdx.x;
  const int wave = tid >> 5, lane = tid & 31;
  const int l = lane & 15, half = lane >> 4;
  const int bh = blockIdx.y, b = bh >> 4, h = bh & 15;
  const int t0 = blockIdx.x * 64 + wave * 16;

  // this thread's fixed chunk coordinates for K/V staging (16B chunks)
  const int sr0 = tid >> 3,  ch0 = (tid & 7) * 8;          // chunk A (rows 0-15)
  const int sr1 = sr0 + 16;                                 // chunk B (rows 16-31)

  // Q A-fragments (persist in VGPRs all kernel; DH=64 -> two K-steps)
  const _Float16* aq = qs + ((size_t)(b * T + t0)) * D + h * DH + (size_t)l * D + 8 * half;
  const v16h a0 = load_a16(aq);
  const v16h a1 = load_a16(aq + 32);

  float Mrow[8], Lrow[8];
  v8f O[4] = {v8f{}, v8f{}, v8f{}, v8f{}};
#pragma unroll
  for (int r = 0; r < 8; ++r) { Mrow[r] = -3.0e38f; Lrow[r] = 0.f; }

  const unsigned char* mbase = mask + (size_t)b * T * T;
  const float* kssr = kssp + (size_t)bh * T;
  const _Float16* kbase = ks + ((size_t)b * T) * D + h * DH;
  const _Float16* vbase = vv + ((size_t)b * T) * D + h * DH;

  // ---- prologue: async-stage K block 0 into Ks[0] ----
  async_b128_to_lds(kbase + (size_t)sr0 * D + ch0, &Ks[0][sr0 * 64 + ch0]);
  async_b128_to_lds(kbase + (size_t)sr1 * D + ch0, &Ks[0][sr1 * 64 + ch0]);

  for (int s0 = 0; s0 < T; s0 += 32) {
    const int buf = (s0 >> 5) & 1;
    wait_async0();                       // my K(s0) chunks resident in LDS
    __syncthreads();                     // => everyone's chunks resident

    // ---- K fragments for S = Q*K^T (two 16x16 s-tiles) ----
    const _Float16* Kb = Ks[buf];
    v16h bk0a = load_a16(&Kb[l * 64 + 8 * half]);
    v16h bk0b = load_a16(&Kb[l * 64 + 32 + 8 * half]);
    v16h bk1a = load_a16(&Kb[(16 + l) * 64 + 8 * half]);
    v16h bk1b = load_a16(&Kb[(16 + l) * 64 + 32 + 8 * half]);

    // ---- issue async copy of NEXT K block (overlaps softmax + PV below) --
    if (s0 + 32 < T) {
      const _Float16* kn = kbase + (size_t)(s0 + 32) * D;
      async_b128_to_lds(kn + (size_t)sr0 * D + ch0, &Ks[buf ^ 1][sr0 * 64 + ch0]);
      async_b128_to_lds(kn + (size_t)sr1 * D + ch0, &Ks[buf ^ 1][sr1 * 64 + ch0]);
    }

    // ---- stage V (transposed) synchronously ----
#pragma unroll
    for (int it = 0; it < 2; ++it) {
      int sr = sr0 + it * 16;
      v8h vt = *(const v8h*)(vbase + (size_t)(s0 + sr) * D + ch0);
#pragma unroll
      for (int e = 0; e < 8; ++e) Vt[(ch0 + e) * 32 + sr] = vt[e];
    }
    // prefetch next V block (global_prefetch_b8)
    if (s0 + 32 < T)
      __builtin_prefetch(vbase + (size_t)(s0 + 32 + sr0) * D + ch0, 0, 1);
    __syncthreads();                     // Vt ready for all waves

    v8f c0 = {}; c0 = wmma_f16(a0, bk0a, c0); c0 = wmma_f16(a1, bk0b, c0);
    v8f c1 = {}; c1 = wmma_f16(a0, bk1a, c1); c1 = wmma_f16(a1, bk1b, c1);

    const float kv0 = kssr[s0 + l];
    const float kv1 = kssr[s0 + 16 + l];

    float p0[8], p1[8], alpha[8];
#pragma unroll
    for (int r = 0; r < 8; ++r) {
      const int trow = t0 + r + 8 * half;
      const unsigned char* mp = mbase + (size_t)trow * T + s0 + l;
      float sv0 = 0.0625f * c0[r] - 0.03125f * kv0 + (mp[0]  ? 0.f : -1.0e30f);
      float sv1 = 0.0625f * c1[r] - 0.03125f * kv1 + (mp[16] ? 0.f : -1.0e30f);
      // row max across the 16 lanes of this half (C layout: row spans lanes)
      float mx = fmaxf(sv0, sv1);
      mx = fmaxf(mx, __shfl_xor(mx, 1, 16));
      mx = fmaxf(mx, __shfl_xor(mx, 2, 16));
      mx = fmaxf(mx, __shfl_xor(mx, 4, 16));
      mx = fmaxf(mx, __shfl_xor(mx, 8, 16));
      float Mn = fmaxf(Mrow[r], mx);
      float al = __expf(Mrow[r] - Mn);
      float e0 = __expf(sv0 - Mn);
      float e1 = __expf(sv1 - Mn);
      float rs = e0 + e1;
      rs += __shfl_xor(rs, 1, 16);
      rs += __shfl_xor(rs, 2, 16);
      rs += __shfl_xor(rs, 4, 16);
      rs += __shfl_xor(rs, 8, 16);
      Lrow[r] = Lrow[r] * al + rs;
      Mrow[r] = Mn;
      alpha[r] = al;
      p0[r] = e0; p1[r] = e1;
    }

    // rescale running output (rows match alpha indexing by construction)
#pragma unroll
    for (int j = 0; j < 4; ++j)
#pragma unroll
      for (int r = 0; r < 8; ++r) O[j][r] *= alpha[r];

    // ---- transpose P (C layout -> A layout) through per-wave LDS ----
    _Float16* pl = Pl[wave];
#pragma unroll
    for (int r = 0; r < 8; ++r) {
      int M = r + 8 * half;
      pl[M * 32 + l]      = (_Float16)p0[r];
      pl[M * 32 + 16 + l] = (_Float16)p1[r];
    }
    // DS ops from the same wave are in-order; loads below see the stores above.
    v8h plo = *(const v8h*)&pl[l * 32 + 8 * half];
    v8h phi = *(const v8h*)&pl[l * 32 + 16 + 8 * half];
    v16h ap = __builtin_shufflevector(plo, phi, 0,1,2,3,4,5,6,7,8,9,10,11,12,13,14,15);

    // ---- O += P * V  (four dh tiles of 16) ----
#pragma unroll
    for (int j = 0; j < 4; ++j) {
      v16h bvf = load_a16(&Vt[(j * 16 + l) * 32 + 8 * half]);
      O[j] = wmma_f16(ap, bvf, O[j]);
    }
  }

  // ---- epilogue: normalize and store f16 [B,T,H*DH] ----
  _Float16* op = ah + ((size_t)(b * T + t0)) * D + h * DH;
#pragma unroll
  for (int j = 0; j < 4; ++j)
#pragma unroll
    for (int r = 0; r < 8; ++r) {
      int M = r + 8 * half;
      op[(size_t)M * D + j * 16 + l] = (_Float16)(O[j][r] / Lrow[r]);
    }
}

// ---------------- kernel 5: output projection (WMMA, f32 out) -------------

__global__ void __launch_bounds__(256)
oproj_kernel(const _Float16* __restrict__ ah, const _Float16* __restrict__ Wot,
             const float* __restrict__ bo, float* __restrict__ out) {
  const int wave = (blockIdx.x * blockDim.x + threadIdx.x) >> 5;
  const int lane = threadIdx.x & 31;
  const int mt = wave / (D / 16), nt = wave % (D / 16);
  v8f c = mm_tile(ah + (size_t)mt * 16 * D, D, Wot + (size_t)nt * 16 * D, D, D, lane);
  const int l = lane & 15, half = lane >> 4;
  const int n = nt * 16 + l;
  const float bb = bo[n];
#pragma unroll
  for (int r = 0; r < 8; ++r) {
    int m = mt * 16 + r + 8 * half;
    out[(size_t)m * D + n] = c[r] + bb;
  }
}

// ---------------- host launcher -------------------------------------------

extern "C" void kernel_launch(void* const* d_in, const int* in_sizes, int n_in,
                              void* d_out, int out_size, void* d_ws, size_t ws_size,
                              hipStream_t stream) {
  (void)in_sizes; (void)n_in; (void)out_size; (void)ws_size;
  const float* x        = (const float*)d_in[0];
  const unsigned char* mask = (const unsigned char*)d_in[1];   // bool -> 1 byte
  const float* Wq = (const float*)d_in[2];
  const float* bq = (const float*)d_in[3];
  const float* Wk = (const float*)d_in[4];
  const float* bk = (const float*)d_in[5];
  const float* Wv = (const float*)d_in[6];
  const float* bv = (const float*)d_in[7];
  const float* Wo = (const float*)d_in[8];
  const float* bo = (const float*)d_in[9];
  const float* log_diag = (const float*)d_in[10];

  char* ws = (char*)d_ws;
  const size_t MB = 1024 * 1024;
  _Float16* xh  = (_Float16*)(ws + 0 * MB);    // 8 MB   x in f16
  _Float16* Wqt = (_Float16*)(ws + 8 * MB);    // 2 MB   transposed f16 weights
  _Float16* Wkt = (_Float16*)(ws + 10 * MB);
  _Float16* Wvt = (_Float16*)(ws + 12 * MB);
  _Float16* Wot = (_Float16*)(ws + 14 * MB);
  _Float16* qsb = (_Float16*)(ws + 16 * MB);   // 8 MB   q * sqrt(diag)
  _Float16* ksb = (_Float16*)(ws + 24 * MB);   // 8 MB   k * sqrt(diag)
  _Float16* vvb = (_Float16*)(ws + 32 * MB);   // 8 MB   v
  _Float16* ahb = (_Float16*)(ws + 40 * MB);   // 8 MB   attention output
  float*    kssb = (float*)(ws + 48 * MB);     // 256 KB |ks|^2
  float*    sdb  = (float*)(ws + 49 * MB);     // 4 KB   sqrt(softplus(diag)+eps)

  prep_kernel<<<(Bq * T * D) / 256, 256, 0, stream>>>(
      x, Wq, Wk, Wv, Wo, log_diag, xh, Wqt, Wkt, Wvt, Wot, sdb);

  // 3 matrices * (256 m-tiles * 64 n-tiles) waves, 8 waves/block
  qkv_kernel<<<(3 * (Bq * T / 16) * (D / 16)) / 8, 256, 0, stream>>>(
      xh, Wqt, Wkt, Wvt, bq, bk, bv, sdb, qsb, ksb, vvb);

  kss_kernel<<<(Bq * H * T) / 256, 256, 0, stream>>>(ksb, kssb);

  attn_kernel<<<dim3(T / 64, Bq * H), 128, 0, stream>>>(
      qsb, ksb, vvb, kssb, mask, ahb);

  oproj_kernel<<<((Bq * T / 16) * (D / 16)) / 8, 256, 0, stream>>>(
      ahb, Wot, bo, (float*)d_out);
}
